// LSTM_Model_85177791414678
// MI455X (gfx1250) — matrix operand order repaired and checked by reading, not compiled, not run on hardware
//
#include <hip/hip_runtime.h>
#include <hip/hip_bf16.h>
#include <math.h>

typedef __attribute__((ext_vector_type(16))) _Float16 v16h;
typedef __attribute__((ext_vector_type(8)))  _Float16 v8h;
typedef __attribute__((ext_vector_type(8)))  float    v8f;

#define B_SZ   16384
#define T_SZ   40
#define E_SZ   50
#define H_SZ   50
#define G_SZ   200      // 4H
#define C_SZ   2
#define KP     128      // padded K: [xe 0..49 | h 50..99 | zero pad 100..127]
#define NP     208      // 4H padded to 13*16
#define NTILES 13
#define WPB    4        // waves per block
#define ROWS   16       // batch rows per wave (WMMA M)

__launch_bounds__(32*WPB, 1)
__global__ void lstm_fused_kernel(
    const int*   __restrict__ x,
    const float* __restrict__ h0,
    const float* __restrict__ c0,
    const float* __restrict__ emb,
    const float* __restrict__ W_ih,
    const float* __restrict__ W_hh,
    const float* __restrict__ b_ih,
    const float* __restrict__ b_hh,
    const float* __restrict__ W_lin,
    const float* __restrict__ b_lin,
    float*       __restrict__ out)
{
    // Combined weight Wc[n][k]: k<50 -> W_ih[n][k], 50<=k<100 -> W_hh[n][k-50], else 0.
    __shared__ __align__(16) _Float16 sW[NP][KP];               // 53 KB, shared by all waves
    __shared__ float   sBias[NP];                               // b_ih + b_hh (padded)
    __shared__ __align__(16) _Float16 sA[WPB][ROWS][KP];        // per-wave A tile (f16)
    __shared__ float   sG[WPB][ROWS][NP];                       // per-wave gate pre-activations
    __shared__ float   sC[WPB][ROWS][H_SZ];                     // cell state
    __shared__ float   sH[WPB][ROWS][H_SZ];                     // hidden state (f32 master copy)

    const int tid   = threadIdx.x;
    const int wave  = tid >> 5;
    const int lane  = tid & 31;
    const int hb    = lane >> 4;    // lane half (K-half for A/B, M-half for D)
    const int nl    = lane & 15;    // M for A-frag, N for B/D frags
    const int Rbase = blockIdx.x * (WPB*ROWS) + wave * ROWS;

    // ---- one-time: weights (f32 -> f16) + fused bias into LDS ----
    for (int i = tid; i < NP*KP; i += WPB*32) {
        int n = i >> 7;             // KP == 128
        int k = i & (KP-1);
        float v = 0.f;
        if (n < G_SZ) {
            if (k < H_SZ)        v = W_ih[n*E_SZ + k];
            else if (k < 2*H_SZ) v = W_hh[n*H_SZ + (k - H_SZ)];
        }
        sW[n][k] = (_Float16)v;
    }
    for (int i = tid; i < NP; i += WPB*32)
        sBias[i] = (i < G_SZ) ? (b_ih[i] + b_hh[i]) : 0.f;

    // ---- per-wave state init (reference passes h0=c0=0, but read them anyway) ----
    for (int idx = lane; idx < ROWS*H_SZ; idx += 32) {
        int m = idx / H_SZ, j = idx % H_SZ;
        sH[wave][m][j] = h0[(Rbase+m)*H_SZ + j];
        sC[wave][m][j] = c0[(Rbase+m)*H_SZ + j];
    }
    for (int idx = lane; idx < ROWS*(KP - 2*H_SZ); idx += 32) {
        int m = idx / (KP - 2*H_SZ), k = idx % (KP - 2*H_SZ);
        sA[wave][m][2*H_SZ + k] = (_Float16)0.f;   // zero K-pad, constant across steps
    }
    __syncthreads();

    for (int t = 0; t < T_SZ; ++t) {
        // ---- phase 1: stage A = [emb[x[b,t]] | h] as f16 ----
        for (int m = 0; m < ROWS; ++m) {
            int tok = x[(Rbase + m)*T_SZ + t];
            const float* er = emb + (long)tok * E_SZ;
            for (int k = lane; k < 2*H_SZ; k += 32) {
                float v = (k < E_SZ) ? er[k] : sH[wave][m][k - H_SZ];
                sA[wave][m][k] = (_Float16)v;
            }
        }
        __syncthreads();

        // ---- phase 2: g[16x200] = A[16x128] * Wc^T via v_wmma_f32_16x16x32_f16 ----
        // A frag (16-bit A 16x32): lane m=nl; elem j -> K = 32*kt + 16*hb + j
        v16h afrag[4];
        for (int kt = 0; kt < 4; ++kt) {
            const _Float16* arow = &sA[wave][nl][kt*32 + hb*16];
            v8h lo = *(const v8h*)(arow);
            v8h hi = *(const v8h*)(arow + 8);
            afrag[kt] = __builtin_shufflevector(lo, hi,
                         0,1,2,3,4,5,6,7,8,9,10,11,12,13,14,15);
        }
        for (int nt = 0; nt < NTILES; ++nt) {
            v8f acc = {};
            // B frag (32x16): lane n=nl, K-half=hb; elem j -> K = 32*kt + 16*hb + j
            // B[k][n] = Wc[n][k] -> 16 contiguous f16 from sW row
            const _Float16* wrow = &sW[nt*16 + nl][hb*16];
            for (int kt = 0; kt < 4; ++kt) {
                v16h b = *(const v16h*)(wrow + kt*32);
                acc = __builtin_amdgcn_wmma_f32_16x16x32_f16(
                          false, afrag[kt], false, b, (short)0, acc, false, false);
            }
            // D frag: elem r holds (M = r + 8*hb, N = 16*nt + nl)
            for (int r = 0; r < 8; ++r)
                sG[wave][hb*8 + r][nt*16 + nl] = acc[r];
        }
        __syncthreads();

        // ---- phase 3: gates + state update (i,f,g,o order, PyTorch convention) ----
        for (int q = 0; q < (ROWS*H_SZ)/32; ++q) {        // 800/32 = 25 per lane
            int idx = q*32 + lane;
            int m = idx / H_SZ, j = idx % H_SZ;
            float gi = sG[wave][m][j          ] + sBias[j          ];
            float gf = sG[wave][m][j +   H_SZ ] + sBias[j +   H_SZ ];
            float gg = sG[wave][m][j + 2*H_SZ ] + sBias[j + 2*H_SZ ];
            float go = sG[wave][m][j + 3*H_SZ ] + sBias[j + 3*H_SZ ];
            float i_ = 1.f / (1.f + __expf(-gi));
            float f_ = 1.f / (1.f + __expf(-gf));
            float g_ = tanhf(gg);
            float o_ = 1.f / (1.f + __expf(-go));
            float c_ = f_ * sC[wave][m][j] + i_ * g_;
            sC[wave][m][j] = c_;
            sH[wave][m][j] = o_ * tanhf(c_);
        }
        __syncthreads();
    }

    // ---- head: out[b][cc] = hT . W_lin[cc] + b_lin[cc]  (16 rows x 2 classes = 32 lanes)
    {
        int m = nl, cc = hb;
        float acc = b_lin[cc];
        for (int j = 0; j < H_SZ; ++j)
            acc += sH[wave][m][j] * W_lin[cc*H_SZ + j];
        out[(Rbase + m)*C_SZ + cc] = acc;
    }
}

extern "C" void kernel_launch(void* const* d_in, const int* in_sizes, int n_in,
                              void* d_out, int out_size, void* d_ws, size_t ws_size,
                              hipStream_t stream) {
    (void)in_sizes; (void)n_in; (void)out_size; (void)d_ws; (void)ws_size;
    const int*   x     = (const int*)  d_in[0];
    const float* h0    = (const float*)d_in[1];
    const float* c0    = (const float*)d_in[2];
    const float* emb   = (const float*)d_in[3];
    const float* W_ih  = (const float*)d_in[4];
    const float* W_hh  = (const float*)d_in[5];
    const float* b_ih  = (const float*)d_in[6];
    const float* b_hh  = (const float*)d_in[7];
    const float* W_lin = (const float*)d_in[8];
    const float* b_lin = (const float*)d_in[9];
    float* out = (float*)d_out;

    dim3 grid(B_SZ / (WPB*ROWS));   // 256 blocks x 4 waves x 16 rows = 16384
    dim3 block(WPB*32);             // 128 threads = 4 wave32s
    hipLaunchKernelGGL(lstm_fused_kernel, grid, block, 0, stream,
                       x, h0, c0, emb, W_ih, W_hh, b_ih, b_hh, W_lin, b_lin, out);
}
